// GlobalSpatialAttention_76132590289444
// MI455X (gfx1250) — compile-verified, matrix operand order checked by reading, not run on hardware
//
#include <hip/hip_runtime.h>
#include <hip/hip_bf16.h>
#include <math.h>

// ---------------------------------------------------------------------------
// GlobalSpatialAttention for MI455X (gfx1250, wave32, WMMA).
// B=16, C=256, H=W=64, HID=128 (gate width 512), EQ=256.
// All GEMM-like work (LSTM input projection + recurrence, conv_right) runs on
// v_wmma_f32_16x16x32_f16. conv_left path is algebraically collapsed to a
// 16x256x256 GEMM on the spatial mean (mean of linear == linear of mean).
// ---------------------------------------------------------------------------

typedef __attribute__((ext_vector_type(16))) _Float16 v16h;
typedef __attribute__((ext_vector_type(8)))  _Float16 v8h;
typedef __attribute__((ext_vector_type(8)))  float    v8f;

#define DEV __device__ __forceinline__

DEV float sigf(float x) { return 1.0f / (1.0f + __expf(-x)); }

// Load a 16x32 f16 WMMA operand fragment from a row-major matrix.
// CDNA5 layout (ISA 7.12.2): lane m = lane&15 selects the row; lanes 0-15
// hold K = {0..7, 16..23}, lanes 16-31 hold K = {8..15, 24..31}. Both halves
// are contiguous 16-byte runs -> two b128 loads per lane. Works for A (rows of
// the activation) and for B when the weight is stored PyTorch-style [N][K]
// (we compute A @ W^T, so B's "row" n is contiguous in W).
DEV v16h load_frag(const _Float16* __restrict__ base, int ld, int lane) {
  const int m    = lane & 15;
  const int koff = (lane & 16) ? 8 : 0;
  const _Float16* p = base + (size_t)m * ld + koff;
  v8h lo = *(const v8h*)(p);
  v8h hi = *(const v8h*)(p + 16);
  v16h r;
#pragma unroll
  for (int i = 0; i < 8; ++i) { r[i] = lo[i]; r[i + 8] = hi[i]; }
  return r;
}

// ---------------------------------------------------------------------------
// Fused bidirectional LSTM layer.
//   x   : f16 [1024 sequences][64 timesteps][256 features], row = n*64 + t
//   out : f16 [65536 rows][256], forward dir -> cols 0..127, backward 128..255
//         out row for (seq n, time t) = ((n>>6)*64 + t)*64 + (n&63)
//         (this is exactly the layout the *next* stage consumes).
// Each block owns 16 sequences; 8 waves each produce 64 gate columns per step
// (4 N-tiles x (8 input + 4 recurrent) k-steps = 48 WMMA/wave/step).
// ---------------------------------------------------------------------------
__global__ __launch_bounds__(256)
void bilstm_wmma(const _Float16* __restrict__ x,
                 const _Float16* __restrict__ WihF, const _Float16* __restrict__ WhhF,
                 const float* __restrict__ bF,
                 const _Float16* __restrict__ WihB, const _Float16* __restrict__ WhhB,
                 const float* __restrict__ bB,
                 _Float16* __restrict__ out) {
  const int dir  = blockIdx.y;
  const int seq0 = blockIdx.x * 16;
  const int tid  = threadIdx.x;
  const int lane = tid & 31;
  const int wid  = tid >> 5;

  const _Float16* Wih  = dir ? WihB : WihF;   // [512][256]
  const _Float16* Whh  = dir ? WhhB : WhhF;   // [512][128]
  const float*    bias = dir ? bB   : bF;     // [512]

  __shared__ __align__(16) _Float16 sh_h[16 * 128];  // h_{t-1}, f16
  __shared__ float sh_g[16 * 512];                   // gate pre-activations

  float c[8];
#pragma unroll
  for (int p = 0; p < 8; ++p) c[p] = 0.0f;
  for (int e = tid; e < 16 * 128; e += 256) sh_h[e] = (_Float16)0.0f;
  __syncthreads();

  const int colBase = wid * 64;

  for (int s = 0; s < 64; ++s) {
    const int t = dir ? (63 - s) : s;

    v8f acc[4];
#pragma unroll
    for (int nt = 0; nt < 4; ++nt) {
      const float bv = bias[colBase + nt * 16 + (lane & 15)];
#pragma unroll
      for (int r = 0; r < 8; ++r) acc[nt][r] = bv;
    }

    // ---- input projection: x_t[16x256] @ Wih^T, K = 256 -> 8 k-steps ----
    const _Float16* xbase = x + ((size_t)seq0 * 64 + t) * 256;
    if (s + 1 < 64) {  // global_prefetch_b8 the next timestep's activations
      const int tn = dir ? (62 - s) : (s + 1);
      __builtin_prefetch(x + ((size_t)seq0 * 64 + tn) * 256 + (size_t)lane * 64, 0, 1);
    }
#pragma unroll
    for (int ks = 0; ks < 8; ++ks) {
      v16h a = load_frag(xbase + ks * 32, 64 * 256, lane);
#pragma unroll
      for (int nt = 0; nt < 4; ++nt) {
        v16h b = load_frag(Wih + (size_t)(colBase + nt * 16) * 256 + ks * 32, 256, lane);
        acc[nt] = __builtin_amdgcn_wmma_f32_16x16x32_f16(
            false, a, false, b, (short)0, acc[nt], false, false);
      }
    }
    // ---- recurrence: h_{t-1}[16x128] @ Whh^T, K = 128 -> 4 k-steps ----
#pragma unroll
    for (int ks = 0; ks < 4; ++ks) {
      v16h a = load_frag((const _Float16*)sh_h + ks * 32, 128, lane);
#pragma unroll
      for (int nt = 0; nt < 4; ++nt) {
        v16h b = load_frag(Whh + (size_t)(colBase + nt * 16) * 128 + ks * 32, 128, lane);
        acc[nt] = __builtin_amdgcn_wmma_f32_16x16x32_f16(
            false, a, false, b, (short)0, acc[nt], false, false);
      }
    }

    // Stage gates to LDS (acc layout: lane&15 = column, VGPR r = row m).
    const int mhi = (lane & 16) ? 8 : 0;
#pragma unroll
    for (int nt = 0; nt < 4; ++nt) {
      const int col = colBase + nt * 16 + (lane & 15);
#pragma unroll
      for (int r = 0; r < 8; ++r) sh_g[(r + mhi) * 512 + col] = acc[nt][r];
    }
    __syncthreads();

    // Elementwise LSTM cell: 2048 (seq,hid) pairs, 8 per thread; c in regs.
#pragma unroll
    for (int p = 0; p < 8; ++p) {
      const int e  = tid * 8 + p;
      const int sq = e >> 7;
      const int j  = e & 127;
      const float gi = sh_g[sq * 512 + j];
      const float gf = sh_g[sq * 512 + 128 + j];
      const float gg = sh_g[sq * 512 + 256 + j];
      const float go = sh_g[sq * 512 + 384 + j];
      const float cn = sigf(gf) * c[p] + sigf(gi) * tanhf(gg);
      const float hn = sigf(go) * tanhf(cn);
      c[p] = cn;
      sh_h[sq * 128 + j] = (_Float16)hn;
      const int n    = seq0 + sq;
      const int orow = ((n >> 6) * 64 + t) * 64 + (n & 63);
      out[(size_t)orow * 256 + dir * 128 + j] = (_Float16)hn;
    }
    __syncthreads();
  }
}

// ---------------------------------------------------------------------------
// conv_right as WMMA GEMM: vf[b][o][s] = sum_c ver[row(b,s)][c] * Wr[o][c]
// rows r = (b*64 + w)*64 + h, spatial s = r & 4095 (== w*64 + h), b = r>>12.
// ---------------------------------------------------------------------------
__global__ __launch_bounds__(256)
void gemm_vf_wmma(const _Float16* __restrict__ A,     // [65536][256]
                  const _Float16* __restrict__ Wr16,  // [256][256]
                  float* __restrict__ vf) {           // [16][256][4096]
  const int tid = threadIdx.x, lane = tid & 31, wid = tid >> 5;
  const int r0 = blockIdx.x * 16;
  v8f acc[2];
#pragma unroll
  for (int nt = 0; nt < 2; ++nt)
#pragma unroll
    for (int r = 0; r < 8; ++r) acc[nt][r] = 0.0f;

  const _Float16* abase = A + (size_t)r0 * 256;
#pragma unroll
  for (int ks = 0; ks < 8; ++ks) {
    v16h a = load_frag(abase + ks * 32, 256, lane);
#pragma unroll
    for (int nt = 0; nt < 2; ++nt) {
      const int col = wid * 32 + nt * 16;
      v16h b = load_frag(Wr16 + (size_t)col * 256 + ks * 32, 256, lane);
      acc[nt] = __builtin_amdgcn_wmma_f32_16x16x32_f16(
          false, a, false, b, (short)0, acc[nt], false, false);
    }
  }
  const int mhi = (lane & 16) ? 8 : 0;
#pragma unroll
  for (int nt = 0; nt < 2; ++nt) {
    const int col = wid * 32 + nt * 16 + (lane & 15);
#pragma unroll
    for (int r = 0; r < 8; ++r) {
      const int row = r0 + r + mhi;
      const int b = row >> 12, s = row & 4095;
      vf[(size_t)b * 1048576 + (size_t)col * 4096 + s] = acc[nt][r];
    }
  }
}

// ------------------------------- small kernels ------------------------------

__global__ void f32_to_f16_k(const float* __restrict__ in, _Float16* __restrict__ out, int n) {
  const int i = blockIdx.x * 256 + threadIdx.x;
  if (i < n) out[i] = (_Float16)in[i];
}

// xh16[((b*64+w)*64+h)*256 + c] = input[b][c][h][w]  (f32 -> f16)
__global__ void transpose_in_k(const float* __restrict__ in, _Float16* __restrict__ out) {
  const int o   = blockIdx.x * 256 + threadIdx.x;  // 16.7M
  const int cc  = o & 255;
  const int row = o >> 8;
  const int h = row & 63, w = (row >> 6) & 63, b = row >> 12;
  out[o] = (_Float16)in[((size_t)(b * 256 + cc) * 64 + h) * 64 + w];
}

__global__ void mean_k(const float* __restrict__ in, float* __restrict__ mean) {
  __shared__ float red[256];
  const int bc = blockIdx.x;  // b*256 + c
  const float* p = in + (size_t)bc * 4096;
  float s = 0.0f;
  for (int i = threadIdx.x; i < 4096; i += 256) s += p[i];
  red[threadIdx.x] = s; __syncthreads();
  for (int st = 128; st > 0; st >>= 1) {
    if (threadIdx.x < st) red[threadIdx.x] += red[threadIdx.x + st];
    __syncthreads();
  }
  if (threadIdx.x == 0) mean[bc] = red[0] * (1.0f / 4096.0f);
}

// q_avg[b][o] = softmax_o( Wl[o][:] . mean[b][:] )
__global__ void qavg_k(const float* __restrict__ mean, const float* __restrict__ Wl,
                       float* __restrict__ qavg) {
  __shared__ float sm[256];
  __shared__ float red[256];
  const int b = blockIdx.x, o = threadIdx.x;
  sm[o] = mean[b * 256 + o];
  __syncthreads();
  float v = 0.0f;
  const float* wr = Wl + (size_t)o * 256;
  for (int cc = 0; cc < 256; ++cc) v += sm[cc] * wr[cc];
  red[o] = v; __syncthreads();
  for (int st = 128; st > 0; st >>= 1) {
    if (o < st) red[o] = fmaxf(red[o], red[o + st]);
    __syncthreads();
  }
  const float mx = red[0]; __syncthreads();
  const float e = __expf(v - mx);
  red[o] = e; __syncthreads();
  for (int st = 128; st > 0; st >>= 1) {
    if (o < st) red[o] += red[o + st];
    __syncthreads();
  }
  qavg[b * 256 + o] = e / red[0];
}

// per-(b,o) spatial softmax stats over 4096 elements
__global__ void softstat_k(const float* __restrict__ vf, float* __restrict__ mx,
                           float* __restrict__ sm) {
  __shared__ float red[256];
  const int bo = blockIdx.x;
  const float* p = vf + (size_t)bo * 4096;
  float m = -1e30f;
  for (int i = threadIdx.x; i < 4096; i += 256) m = fmaxf(m, p[i]);
  red[threadIdx.x] = m; __syncthreads();
  for (int st = 128; st > 0; st >>= 1) {
    if (threadIdx.x < st) red[threadIdx.x] = fmaxf(red[threadIdx.x], red[threadIdx.x + st]);
    __syncthreads();
  }
  m = red[0]; __syncthreads();
  float s = 0.0f;
  for (int i = threadIdx.x; i < 4096; i += 256) s += __expf(p[i] - m);
  red[threadIdx.x] = s; __syncthreads();
  for (int st = 128; st > 0; st >>= 1) {
    if (threadIdx.x < st) red[threadIdx.x] += red[threadIdx.x + st];
    __syncthreads();
  }
  if (threadIdx.x == 0) { mx[bo] = m; sm[bo] = red[0]; }
}

// att[b][idx] = sum_o (q_avg[b][o]/sumexp[b][o]) * exp(vf[b][o][idx] - max[b][o])
__global__ void att_k(const float* __restrict__ vf, const float* __restrict__ qavg,
                      const float* __restrict__ mx, const float* __restrict__ sm,
                      float* __restrict__ att) {
  __shared__ float qn[256], qm[256];
  const int b   = blockIdx.x >> 4;
  const int idx = (blockIdx.x & 15) * 256 + threadIdx.x;
  qn[threadIdx.x] = qavg[b * 256 + threadIdx.x] / sm[b * 256 + threadIdx.x];
  qm[threadIdx.x] = mx[b * 256 + threadIdx.x];
  __syncthreads();
  const float* p = vf + (size_t)b * 1048576 + idx;
  float a = 0.0f;
  for (int o = 0; o < 256; ++o) a += qn[o] * __expf(p[(size_t)o * 4096] - qm[o]);
  att[b * 4096 + idx] = a;
}

// out = in * sigmoid(att[h*64+w]) * alpha + in
__global__ void final_k(const float* __restrict__ in, const float* __restrict__ att,
                        const float* __restrict__ alpha, float* __restrict__ out) {
  const size_t i = (size_t)blockIdx.x * 256 + threadIdx.x;  // 16.7M
  const int w = (int)(i & 63), h = (int)((i >> 6) & 63);
  const int b = (int)(i >> 20);
  const float a = sigf(att[b * 4096 + h * 64 + w]);
  const float x = in[i];
  out[i] = x * a * alpha[0] + x;
}

// ------------------------------ workspace map -------------------------------
static constexpr size_t NPIX    = 16777216ull;  // 16*64*64*256 elements
static constexpr size_t OFF_XH  = 0;                    // f16, 32 MB
static constexpr size_t OFF_HOR = OFF_XH  + NPIX * 2;   // f16, 32 MB
static constexpr size_t OFF_VER = OFF_HOR + NPIX * 2;   // f16, 32 MB
static constexpr size_t OFF_VF  = OFF_VER + NPIX * 2;   // f32, 64 MB
static constexpr size_t OFF_W   = OFF_VF  + NPIX * 4;   // f16 weight pool
static constexpr size_t OFF_SM  = OFF_W + 851968ull * 2;// small f32 arrays

extern "C" void kernel_launch(void* const* d_in, const int* in_sizes, int n_in,
                              void* d_out, int out_size, void* d_ws, size_t ws_size,
                              hipStream_t stream) {
  const float* input = (const float*)d_in[0];
  const float* hWihF = (const float*)d_in[1];
  const float* hWhhF = (const float*)d_in[2];
  const float* hbF   = (const float*)d_in[3];
  const float* hWihB = (const float*)d_in[4];
  const float* hWhhB = (const float*)d_in[5];
  const float* hbB   = (const float*)d_in[6];
  const float* vWihF = (const float*)d_in[7];
  const float* vWhhF = (const float*)d_in[8];
  const float* vbF   = (const float*)d_in[9];
  const float* vWihB = (const float*)d_in[10];
  const float* vWhhB = (const float*)d_in[11];
  const float* vbB   = (const float*)d_in[12];
  const float* Wr    = (const float*)d_in[13];
  const float* Wl    = (const float*)d_in[14];
  const float* alpha = (const float*)d_in[15];
  float* out = (float*)d_out;

  char* ws = (char*)d_ws;
  _Float16* xh16 = (_Float16*)(ws + OFF_XH);
  _Float16* horb = (_Float16*)(ws + OFF_HOR);
  _Float16* verb = (_Float16*)(ws + OFF_VER);
  float*    vfb  = (float*)(ws + OFF_VF);

  _Float16* wpool   = (_Float16*)(ws + OFF_W);
  _Float16* hWihF16 = wpool;            _Float16* hWihB16 = wpool + 131072;
  _Float16* hWhhF16 = wpool + 262144;   _Float16* hWhhB16 = wpool + 327680;
  _Float16* vWihF16 = wpool + 393216;   _Float16* vWihB16 = wpool + 524288;
  _Float16* vWhhF16 = wpool + 655360;   _Float16* vWhhB16 = wpool + 720896;
  _Float16* Wr16    = wpool + 786432;

  float* meanb = (float*)(ws + OFF_SM);
  float* qavgb = meanb + 4096;
  float* mxb   = meanb + 8192;
  float* smb   = meanb + 12288;
  float* attb  = meanb + 16384;

  // weights -> f16
  f32_to_f16_k<<<512, 256, 0, stream>>>(hWihF, hWihF16, 131072);
  f32_to_f16_k<<<512, 256, 0, stream>>>(hWihB, hWihB16, 131072);
  f32_to_f16_k<<<256, 256, 0, stream>>>(hWhhF, hWhhF16, 65536);
  f32_to_f16_k<<<256, 256, 0, stream>>>(hWhhB, hWhhB16, 65536);
  f32_to_f16_k<<<512, 256, 0, stream>>>(vWihF, vWihF16, 131072);
  f32_to_f16_k<<<512, 256, 0, stream>>>(vWihB, vWihB16, 131072);
  f32_to_f16_k<<<256, 256, 0, stream>>>(vWhhF, vWhhF16, 65536);
  f32_to_f16_k<<<256, 256, 0, stream>>>(vWhhB, vWhhB16, 65536);
  f32_to_f16_k<<<256, 256, 0, stream>>>(Wr, Wr16, 65536);

  // left branch (collapsed conv_left) + horizontal LSTM input layout
  transpose_in_k<<<65536, 256, 0, stream>>>(input, xh16);
  mean_k<<<4096, 256, 0, stream>>>(input, meanb);
  qavg_k<<<16, 256, 0, stream>>>(meanb, Wl, qavgb);

  // horizontal biLSTM (seq = (b,w), time = h) -> horb in vertical layout
  bilstm_wmma<<<dim3(64, 2), 256, 0, stream>>>(
      xh16, hWihF16, hWhhF16, hbF, hWihB16, hWhhB16, hbB, horb);
  // vertical biLSTM (seq = (b,h), time = w) -> verb in conv_right layout
  bilstm_wmma<<<dim3(64, 2), 256, 0, stream>>>(
      horb, vWihF16, vWhhF16, vbF, vWihB16, vWhhB16, vbB, verb);

  // conv_right + spatial softmax + attention + output
  gemm_vf_wmma<<<4096, 256, 0, stream>>>(verb, Wr16, vfb);
  softstat_k<<<4096, 256, 0, stream>>>(vfb, mxb, smb);
  att_k<<<256, 256, 0, stream>>>(vfb, qavgb, mxb, smb, attb);
  final_k<<<65536, 256, 0, stream>>>(input, attb, alpha, out);
}
